// DU_GSB_LSTM_25417616458228
// MI455X (gfx1250) — matrix operand order repaired and checked by reading, not compiled, not run on hardware
//
#include <hip/hip_runtime.h>
#include <math.h>

typedef __attribute__((ext_vector_type(16))) _Float16 v16h;
typedef __attribute__((ext_vector_type(8)))  float    v8f;

#define D_N     512
#define NBATCH  2048
#define NHT     256
#define HIDN    64
#define TSTEPS  64

// ---- workspace layout (float offsets, all 256B aligned) ----
constexpr size_t OFF_HT  = 0;        // 65536  Ht [256x256]
constexpr size_t OFF_A   = 65536;    // 65536  A = HtHt^T + lam I
constexpr size_t OFF_INV = 131072;   // 65536  U = inv(A)/lam
constexpr size_t OFF_WM  = 196608;   // 65536  W = U @ Ht
constexpr size_t OFF_G   = 262144;   // 65536  G = Ht^T @ W
constexpr size_t OFF_J   = 327680;   // 262144 J f32 [512x512]
constexpr size_t OFF_X   = 589824;   // 1048576 X  [b][n]
constexpr size_t OFF_YV  = 1638400;  // 1048576 Yv [b][n]
constexpr size_t OFF_VM  = 2686976;  // 1048576 Vm [b][n]
constexpr size_t OFF_AX  = 3735552;  // 1048576 AxT [b][n] = (J@x)^T
constexpr size_t OFF_HID = 4784128;  // 131072 hidden [b][64]
constexpr size_t OFF_CEL = 4915200;  // 131072 cell   [b][64]
constexpr size_t OFF_HV  = 5046272;  // 512    h vector
constexpr size_t OFF_YT  = 5046784;  // 256    yt
constexpr size_t OFF_UV  = 5047040;  // 256    uvec = U @ z
constexpr size_t OFF_GB  = 5047296;  // 512    gbest
constexpr size_t OFF_E   = 5047808;  // 2048   energy per column
constexpr size_t OFF_GN2 = 5049856;  // 2048   ||gbest-x||^2
constexpr size_t OFF_DGY = 5051904;  // 2048   dot(gd, yv)
constexpr size_t OFF_YN2 = 5053952;  // 2048   ||yv||^2
constexpr size_t OFF_SX  = 5056000;  // 2048   sum x
constexpr size_t OFF_SX2 = 5058048;  // 2048   sum x^2
constexpr size_t OFF_AI  = 5060096;  // 2048   a_i per column
constexpr size_t OFF_SDF = 5062144;  // 2048   soft*decay per column
constexpr size_t OFF_FM  = 5064192;  // 64     feat mean[3], feat std[3]
constexpr size_t OFF_SC  = 5064256;  // 64     scalars: 0=fro 1=c0 2=gbest_e
constexpr size_t OFF_JH  = 5064320;  // 131072 floats -> 262144 halfs (J f16)
constexpr size_t OFF_XH  = 5195392;  // 524288 floats -> 1048576 halfs (x^T f16)
constexpr size_t OFF_SH  = 5719680;  // 524288 floats -> 1048576 halfs (s^T f16)
// end = 6243968 floats = ~25 MB

// ---------- device helpers ----------
__device__ __forceinline__ float sigm(float z) { return 1.0f / (1.0f + expf(-z)); }
__device__ __forceinline__ float swishf(float z) { return z * sigm(z); }
__device__ __forceinline__ float phi_sf(float x) { return swishf(x + 1.f) - swishf(x - 1.f) - 1.f; }
__device__ __forceinline__ float psi_sf(float x) { return sigm(100.f * (fabsf(x) - 1.01f)); }
__device__ __forceinline__ float softplusf(float z) {
    if (z > 20.f) return z;
    return log1pf(expf(z));
}
__device__ __forceinline__ unsigned mix32(unsigned x) {
    x ^= x >> 16; x *= 0x7feb352du;
    x ^= x >> 15; x *= 0x846ca68bu;
    x ^= x >> 16; return x;
}
__device__ __forceinline__ float u01(unsigned x) {
    return (float)(mix32(x) >> 8) * (1.0f / 16777216.0f);
}

__device__ __forceinline__ v8f wmma_f16(v16h a, v16h b, v8f c) {
    // D = A(16x32 f16) * B(32x16 f16) + C(16x16 f32)
    return __builtin_amdgcn_wmma_f32_16x16x32_f16(false, a, false, b, (short)0, c, false, false);
}

// ---------- setup kernels ----------
__global__ void k_build_ht(const float* __restrict__ Hr, const float* __restrict__ Hi,
                           const float* __restrict__ yr, const float* __restrict__ yi,
                           float* __restrict__ Ht, float* __restrict__ yt) {
    int idx = blockIdx.x * 256 + threadIdx.x;
    if (idx >= NHT * NHT) return;
    int r = idx >> 8, c = idx & 255;
    float v;
    if (r < 128) v = (c < 128) ? Hr[r * 128 + c] : -Hi[r * 128 + (c - 128)];
    else         v = (c < 128) ? Hi[(r - 128) * 128 + c] : Hr[(r - 128) * 128 + (c - 128)];
    Ht[idx] = v;
    if (idx < NHT) yt[idx] = (idx < 128) ? yr[idx] : yi[idx - 128];
}

__global__ void k_ata(const float* __restrict__ Ht, const float* __restrict__ lam,
                      float* __restrict__ A) {
    int idx = blockIdx.x * 256 + threadIdx.x;
    if (idx >= NHT * NHT) return;
    int i = idx >> 8, j = idx & 255;
    float acc = 0.f;
    for (int k = 0; k < NHT; ++k) acc += Ht[i * NHT + k] * Ht[j * NHT + k];
    if (i == j) acc += lam[0];
    A[idx] = acc;
}

// single-workgroup Gauss-Jordan inverse of SPD 256x256, then scale by 1/lam
__global__ void k_invert(float* __restrict__ A, float* __restrict__ Inv,
                         const float* __restrict__ lam) {
    const int tid = threadIdx.x;
    for (int t = tid; t < NHT * NHT; t += 1024)
        Inv[t] = ((t >> 8) == (t & 255)) ? 1.f : 0.f;
    __syncthreads();
    __shared__ float colk[NHT];
    __shared__ float rpiv_s;
    for (int k = 0; k < NHT; ++k) {
        if (tid == 0) rpiv_s = 1.f / A[k * NHT + k];
        __syncthreads();
        float rp = rpiv_s;
        if (tid < 256)      A[k * NHT + tid] *= rp;
        else if (tid < 512) Inv[k * NHT + (tid - 256)] *= rp;
        __syncthreads();
        if (tid < 256) colk[tid] = A[tid * NHT + k];
        __syncthreads();
        for (int t = tid; t < 256 * 512; t += 1024) {
            int r = t >> 9, c = t & 511;
            if (r == k) continue;
            float f = colk[r];
            if (c < 256) A[r * NHT + c]          -= f * A[k * NHT + c];
            else         Inv[r * NHT + (c - 256)] -= f * Inv[k * NHT + (c - 256)];
        }
        __syncthreads();
    }
    float il = 1.f / lam[0];
    for (int t = tid; t < NHT * NHT; t += 1024) Inv[t] *= il;
}

// R = P@Q (transA=0) or P^T@Q (transA=1), all 256x256
__global__ void k_gemm256(const float* __restrict__ P, const float* __restrict__ Q,
                          float* __restrict__ R, int transA) {
    int idx = blockIdx.x * 256 + threadIdx.x;
    if (idx >= NHT * NHT) return;
    int i = idx >> 8, j = idx & 255;
    float acc = 0.f;
    if (transA == 0)
        for (int k = 0; k < NHT; ++k) acc += P[i * NHT + k] * Q[k * NHT + j];
    else
        for (int k = 0; k < NHT; ++k) acc += P[k * NHT + i] * Q[k * NHT + j];
    R[idx] = acc;
}

__global__ void k_uvec(const float* __restrict__ U, const float* __restrict__ yt,
                       float* __restrict__ uvec) {
    int i = threadIdx.x;
    const float isq = 0.316227766016838f; // 1/sqrt(qam_var), qam_var=10
    float acc = 0.f;
    for (int k = 0; k < NHT; ++k) acc += U[i * NHT + k] * yt[k] * isq;
    uvec[i] = acc;
}

__global__ void k_hvec(const float* __restrict__ Ht, const float* __restrict__ uvec,
                       float* __restrict__ hvec) {
    int idx = blockIdx.x * 256 + threadIdx.x;
    if (idx >= D_N) return;
    int i = idx & 255, kblk = idx >> 8;
    float sc = (kblk == 0) ? 2.f : 1.f;
    float acc = 0.f;
    for (int r = 0; r < NHT; ++r) acc += Ht[r * NHT + i] * uvec[r];
    hvec[idx] = 2.f * sc * acc;
}

__global__ void k_buildJ(const float* __restrict__ G, float* __restrict__ J,
                         _Float16* __restrict__ Jh, float* __restrict__ fro) {
    int idx = blockIdx.x * 256 + threadIdx.x;
    float val = 0.f;
    if (idx < D_N * D_N) {
        int a = idx >> 9, b = idx & 511;
        int i = a & 255, ka = a >> 8;
        int j = b & 255, kb = b >> 8;
        float sa = (ka == 0) ? 2.f : 1.f;
        float sb = (kb == 0) ? 2.f : 1.f;
        val = -(sa * sb * G[i * NHT + j]) * 0.2f; // 2/qam_var
        if (a == b) val = 0.f;
        J[idx] = val;
        Jh[idx] = (_Float16)val;
    }
    __shared__ float red[256];
    red[threadIdx.x] = val * val;
    __syncthreads();
    for (int s = 128; s > 0; s >>= 1) {
        if (threadIdx.x < s) red[threadIdx.x] += red[threadIdx.x + s];
        __syncthreads();
    }
    if (threadIdx.x == 0) atomicAdd(fro, red[0]);
}

__global__ void k_c0(float* __restrict__ scal) {
    scal[1] = 0.5f * sqrtf(511.0f) / (sqrtf(scal[0]) + 1e-12f);
}

__global__ void k_init(float* __restrict__ X, float* __restrict__ Yv, float* __restrict__ Vm,
                       float* __restrict__ hid, float* __restrict__ cel,
                       const float* __restrict__ h0, const float* __restrict__ c0in,
                       float* __restrict__ scal) {
    int idx = blockIdx.x * 256 + threadIdx.x;
    if (idx < D_N * NBATCH) {
        X[idx]  = 0.02f * (u01((unsigned)idx ^ 0xA511E9B3u) - 0.5f);
        Yv[idx] = 0.02f * (u01((unsigned)idx ^ 0x68BC21EBu) - 0.5f);
        Vm[idx] = 0.f;
    }
    if (idx < NBATCH * HIDN) {
        hid[idx] = h0[idx & 63];
        cel[idx] = c0in[idx & 63];
    }
    if (idx == 0) { scal[0] = 0.f; scal[2] = INFINITY; }
}

// ---------- per-step kernels ----------
// (initial pass only; in-loop this work is fused into k_update)
__global__ void k_conv(const float* __restrict__ X, _Float16* __restrict__ xh,
                       _Float16* __restrict__ sh, float* __restrict__ e) {
    int idx = blockIdx.x * 256 + threadIdx.x;
    if (idx >= D_N * NBATCH) return;
    float x = X[idx];
    float s = phi_sf(x);
    if (s == 0.f) s = 1.f;
    sh[idx] = (_Float16)s;
    xh[idx] = (_Float16)x;
    if (idx < NBATCH) e[idx] = 0.f;
}

union AFrag { v16h v; uint4 q[2]; };

// WMMA GEMM: As = J@s (folded into energy), Ax = J@x (stored).
// block = 256 thr = 8 waves; per wave: 32 J-rows x 16 batch cols, both matrices.
// Per 32-K step: 4 LDS b128 (two A frags) + 4 global b128 (Bs,Bx) -> 4 WMMAs
// (2 loads/WMMA; B fragments each reused across two A fragments).
__global__ void k_gemm_wmma(const _Float16* __restrict__ Jh, const _Float16* __restrict__ sh,
                            const _Float16* __restrict__ xh, const float* __restrict__ hvec,
                            float* __restrict__ AxT, float* __restrict__ e) {
    __shared__ __align__(16) _Float16 Jt[32 * D_N]; // 32 KB J row-tile
    const int tid   = threadIdx.x;
    const int tileM = blockIdx.y; // 32 rows per tile
    {   // contiguous 32 KB row-block of row-major Jh
        const uint4* src = (const uint4*)(Jh + (size_t)tileM * (32 * D_N));
        uint4* dst = (uint4*)Jt;
        for (int t = tid; t < (32 * D_N) / 8; t += 256) dst[t] = src[t];
    }
    __syncthreads();

    const int wave = tid >> 5, lane = tid & 31;
    const int ln = lane & 15, lh = lane >> 4;
    const int bcol = blockIdx.x * 128 + wave * 16 + ln;

    v8f accS0 = {0.f, 0.f, 0.f, 0.f, 0.f, 0.f, 0.f, 0.f};
    v8f accS1 = {0.f, 0.f, 0.f, 0.f, 0.f, 0.f, 0.f, 0.f};
    v8f accX0 = {0.f, 0.f, 0.f, 0.f, 0.f, 0.f, 0.f, 0.f};
    v8f accX1 = {0.f, 0.f, 0.f, 0.f, 0.f, 0.f, 0.f, 0.f};

    const _Float16* jrow0 = Jt + ln * D_N;          // A rows 0..15 of tile
    const _Float16* jrow1 = Jt + (16 + ln) * D_N;   // A rows 16..31 of tile
    const _Float16* srow = sh + (size_t)bcol * D_N; // B column (batch)
    const _Float16* xrow = xh + (size_t)bcol * D_N;
    __builtin_prefetch(srow, 0, 1);
    __builtin_prefetch(xrow, 0, 1);

#pragma unroll 2
    for (int kk = 0; kk < D_N; kk += 32) {
        AFrag a0, a1, bs, bx;
        // A 16x32 f16 layout: lanes0-15 K={0..7,16..23}, lanes16-31 K={8..15,24..31}
        a0.q[0] = *(const uint4*)(jrow0 + kk + lh * 8);
        a0.q[1] = *(const uint4*)(jrow0 + kk + 16 + lh * 8);
        a1.q[0] = *(const uint4*)(jrow1 + kk + lh * 8);
        a1.q[1] = *(const uint4*)(jrow1 + kk + 16 + lh * 8);
        // B 32x16 f16 layout: lane n=l&15, lanes0-15 K={0..7,16..23}, lanes16-31 K={8..15,24..31}
        const uint4* sp_ = (const uint4*)(srow + kk + lh * 8);
        bs.q[0] = sp_[0]; bs.q[1] = sp_[2];
        const uint4* xp_ = (const uint4*)(xrow + kk + lh * 8);
        bx.q[0] = xp_[0]; bx.q[1] = xp_[2];
        accS0 = wmma_f16(a0.v, bs.v, accS0);
        accS1 = wmma_f16(a1.v, bs.v, accS1);
        accX0 = wmma_f16(a0.v, bx.v, accX0);
        accX1 = wmma_f16(a1.v, bx.v, accX1);
    }

    // C layout: VGPR r -> M = r + 8*(lane>>4), N = lane&15
    float contrib = 0.f;
#pragma unroll
    for (int sub = 0; sub < 2; ++sub) {
        const v8f accX = sub ? accX1 : accX0;
        const v8f accS = sub ? accS1 : accS0;
        const int m0 = tileM * 32 + sub * 16 + lh * 8;
        float* axp = AxT + (size_t)bcol * D_N + m0;
        *(float4*)(axp)     = make_float4(accX[0], accX[1], accX[2], accX[3]);
        *(float4*)(axp + 4) = make_float4(accX[4], accX[5], accX[6], accX[7]);
        union { uint4 q; _Float16 hh[8]; } s8;
        s8.q = *(const uint4*)(srow + m0);
#pragma unroll
        for (int r = 0; r < 8; ++r)
            contrib -= (0.5f * accS[r] + hvec[m0 + r]) * (float)s8.hh[r];
    }
    contrib += __shfl_down(contrib, 16, 32);
    if (lane < 16) atomicAdd(&e[bcol], contrib);
}

__global__ void k_argmin(const float* __restrict__ e, const float* __restrict__ X,
                         float* __restrict__ gbest, float* __restrict__ scal) {
    __shared__ float bv[1024];
    __shared__ int   bi_s[1024];
    __shared__ int   flag, sel;
    const int tid = threadIdx.x;
    float v0 = e[tid], v1 = e[tid + 1024];
    float v; int ix;
    if (v1 < v0) { v = v1; ix = tid + 1024; } else { v = v0; ix = tid; }
    bv[tid] = v; bi_s[tid] = ix;
    __syncthreads();
    for (int s = 512; s > 0; s >>= 1) {
        if (tid < s) {
            float ov = bv[tid + s]; int oi = bi_s[tid + s];
            if (ov < bv[tid] || (ov == bv[tid] && oi < bi_s[tid])) { bv[tid] = ov; bi_s[tid] = oi; }
        }
        __syncthreads();
    }
    if (tid == 0) {
        if (bv[0] < scal[2]) { scal[2] = bv[0]; sel = bi_s[0]; flag = 1; }
        else flag = 0;
    }
    __syncthreads();
    if (flag && tid < D_N) gbest[tid] = X[(size_t)sel * D_N + tid];
}

// one wave (32 lanes) per batch column
__global__ void k_stats(const float* __restrict__ X, const float* __restrict__ Yv,
                        const float* __restrict__ gbest,
                        float* __restrict__ gn2, float* __restrict__ dgy, float* __restrict__ yn2,
                        float* __restrict__ sx, float* __restrict__ sx2) {
    const int tid = threadIdx.x;
    const int wave = tid >> 5, lane = tid & 31;
    const int col = blockIdx.x * 8 + wave;
    float a0 = 0.f, a1 = 0.f, a2 = 0.f, a3 = 0.f, a4 = 0.f;
    const float* xc = X + (size_t)col * D_N;
    const float* yc = Yv + (size_t)col * D_N;
    for (int n = lane; n < D_N; n += 32) {
        float x = xc[n], yv = yc[n];
        float g = gbest[n] - x;
        a0 += g * g; a1 += g * yv; a2 += yv * yv; a3 += x; a4 += x * x;
    }
    for (int off = 16; off > 0; off >>= 1) {
        a0 += __shfl_down(a0, off, 32);
        a1 += __shfl_down(a1, off, 32);
        a2 += __shfl_down(a2, off, 32);
        a3 += __shfl_down(a3, off, 32);
        a4 += __shfl_down(a4, off, 32);
    }
    if (lane == 0) {
        gn2[col] = a0; dgy[col] = a1; yn2[col] = a2; sx[col] = a3; sx2[col] = a4;
    }
}

__global__ void k_featstats(const float* __restrict__ e, const float* __restrict__ gn2,
                            const float* __restrict__ sx, const float* __restrict__ sx2,
                            float* __restrict__ fm) {
    __shared__ float red[6][256];
    const int tid = threadIdx.x;
    float s0 = 0.f, s1 = 0.f, s2 = 0.f, q0 = 0.f, q1 = 0.f, q2 = 0.f;
    for (int b = tid; b < NBATCH; b += 256) {
        float f0 = e[b];
        float f1 = sqrtf(gn2[b]);
        float xm = sx[b];
        float f2 = sqrtf(fmaxf(0.f, (sx2[b] - xm * xm / 512.f) / 511.f));
        s0 += f0; q0 += f0 * f0;
        s1 += f1; q1 += f1 * f1;
        s2 += f2; q2 += f2 * f2;
    }
    red[0][tid] = s0; red[1][tid] = s1; red[2][tid] = s2;
    red[3][tid] = q0; red[4][tid] = q1; red[5][tid] = q2;
    __syncthreads();
    for (int s = 128; s > 0; s >>= 1) {
        if (tid < s)
            for (int k = 0; k < 6; ++k) red[k][tid] += red[k][tid + s];
        __syncthreads();
    }
    if (tid == 0) {
        for (int k = 0; k < 3; ++k) {
            float mean = red[k][0] / (float)NBATCH;
            float var = (red[k + 3][0] - red[k][0] * red[k][0] / (float)NBATCH) / (float)(NBATCH - 1);
            fm[k] = mean;
            fm[3 + k] = fmaxf(sqrtf(fmaxf(var, 0.f)), 1e-6f);
        }
    }
}

__global__ void k_lstm(float stepf,
                       const float* __restrict__ e, const float* __restrict__ gn2,
                       const float* __restrict__ dgy, const float* __restrict__ yn2,
                       const float* __restrict__ sx, const float* __restrict__ sx2,
                       const float* __restrict__ fm,
                       float* __restrict__ hid, float* __restrict__ cel,
                       const float* __restrict__ W_ih, const float* __restrict__ W_hh,
                       const float* __restrict__ b_ih, const float* __restrict__ b_hh,
                       const float* __restrict__ W1, const float* __restrict__ b1,
                       const float* __restrict__ W2, const float* __restrict__ b2,
                       float* __restrict__ ai, float* __restrict__ sdf) {
    const int b = blockIdx.x * 256 + threadIdx.x;
    if (b >= NBATCH) return;
    float f0 = e[b];
    float gn = sqrtf(gn2[b]);
    float xm = sx[b];
    float f2 = sqrtf(fmaxf(0.f, (sx2[b] - xm * xm / 512.f) / 511.f));
    float nf0 = (f0 - fm[0]) / fm[3];
    float nf1 = (gn - fm[1]) / fm[4];
    float nf2 = (f2 - fm[2]) / fm[5];

    float hold[HIDN], hnew[HIDN];
    for (int j = 0; j < HIDN; ++j) hold[j] = hid[(size_t)b * HIDN + j];

    for (int j = 0; j < HIDN; ++j) {
        float g4[4];
        for (int gidx = 0; gidx < 4; ++gidx) {
            int r = gidx * HIDN + j; // torch order i,f,g,o
            float acc = b_ih[r] + b_hh[r]
                      + W_ih[r * 3 + 0] * nf0 + W_ih[r * 3 + 1] * nf1 + W_ih[r * 3 + 2] * nf2;
            const float* wr = W_hh + (size_t)r * HIDN;
            for (int k = 0; k < HIDN; ++k) acc += wr[k] * hold[k];
            g4[gidx] = acc;
        }
        float c2 = sigm(g4[1]) * cel[(size_t)b * HIDN + j] + sigm(g4[0]) * tanhf(g4[2]);
        float h2 = sigm(g4[3]) * tanhf(c2);
        cel[(size_t)b * HIDN + j] = c2;
        hnew[j] = h2;
    }
    for (int j = 0; j < HIDN; ++j) hid[(size_t)b * HIDN + j] = hnew[j];

    // MLP: softplus(relu(h@W1^T+b1)@W2^T+b2)
    float z0 = b2[0], z1 = b2[1];
    for (int jj = 0; jj < HIDN; ++jj) {
        float acc = b1[jj];
        const float* wr = W1 + (size_t)jj * HIDN;
        for (int k = 0; k < HIDN; ++k) acc += wr[k] * hnew[k];
        float r1 = fmaxf(acc, 0.f);
        z0 += W2[jj] * r1;
        z1 += W2[HIDN + jj] * r1;
    }
    float a_v = softplusf(z0) - 1.f;
    float b_v = softplusf(z1) - 1.f;

    float yn = sqrtf(yn2[b]);
    float decay = (1.f - stepf) / (1.f + gn);
    const float eps = 1e-8f;
    float cosv = (decay * dgy[b]) / (fmaxf(decay * gn, eps) * fmaxf(yn, eps));
    float soft = tanhf(1000.f * (cosv + b_v));
    ai[b] = a_v;
    sdf[b] = soft * decay;
}

// fused: state update + f16 shadow generation (xh, sh) + energy reset for next step
__global__ void k_update(int iStep, float stepf, float p, float alpha,
                         const float* __restrict__ Delta, const float* __restrict__ eta,
                         const float* __restrict__ scal, const float* __restrict__ hvec,
                         const float* __restrict__ gbest, const float* __restrict__ AxT,
                         const float* __restrict__ ai, const float* __restrict__ sdf,
                         float* __restrict__ X, float* __restrict__ Yv, float* __restrict__ Vm,
                         _Float16* __restrict__ xh, _Float16* __restrict__ sh,
                         float* __restrict__ e) {
    int idx = blockIdx.x * 256 + threadIdx.x;
    if (idx >= D_N * NBATCH) return;
    const int b = idx >> 9, n = idx & 511;
    const float d_i = Delta[iStep], e_i = eta[iStep], c0v = scal[1];
    float x = X[idx], yv = Yv[idx], vm = Vm[idx];
    float gd = gbest[n] - x;
    float ist = sdf[b] * gd;
    vm = alpha * vm + (1.f - alpha) * ist;
    yv += (-(1.f - p) * x + e_i * c0v * (AxT[idx] + hvec[n]) + vm) * d_i;
    x += d_i * yv;
    float u = u01((unsigned)idx ^ (0x9E3779B9u * (unsigned)(iStep + 17)));
    if (u < p) x -= ai[b];
    x = phi_sf(x);
    yv *= (1.f - psi_sf(x));
    X[idx] = x; Yv[idx] = yv; Vm[idx] = vm;
    // f16 shadows for next step's WMMA GEMM (s gets the zero->1 fixup)
    float s = phi_sf(x);
    if (s == 0.f) s = 1.f;
    sh[idx] = (_Float16)s;
    xh[idx] = (_Float16)x;
    if (idx < NBATCH) e[idx] = 0.f;
}

__global__ void k_out(const float* __restrict__ X, float* __restrict__ out) {
    int idx = blockIdx.x * 256 + threadIdx.x;
    if (idx < D_N * NBATCH) out[idx] = X[idx]; // out is x.T = [B][N] == X layout
}

// ---------- host launcher ----------
extern "C" void kernel_launch(void* const* d_in, const int* in_sizes, int n_in,
                              void* d_out, int out_size, void* d_ws, size_t ws_size,
                              hipStream_t stream) {
    (void)in_sizes; (void)n_in; (void)out_size; (void)ws_size;
    const float* Hr    = (const float*)d_in[0];
    const float* Hi    = (const float*)d_in[1];
    const float* yr    = (const float*)d_in[2];
    const float* yi    = (const float*)d_in[3];
    const float* lam   = (const float*)d_in[4];
    const float* Delta = (const float*)d_in[5];
    const float* eta   = (const float*)d_in[6];
    const float* W_ih  = (const float*)d_in[7];
    const float* W_hh  = (const float*)d_in[8];
    const float* b_ih  = (const float*)d_in[9];
    const float* b_hh  = (const float*)d_in[10];
    const float* W1    = (const float*)d_in[11];
    const float* b1    = (const float*)d_in[12];
    const float* W2    = (const float*)d_in[13];
    const float* b2    = (const float*)d_in[14];
    const float* h0    = (const float*)d_in[15];
    const float* c0in  = (const float*)d_in[16];
    // d_in[17] = nbps (=4, hardcoded: M=16, rb=2, D=512)

    float* w = (float*)d_ws;
    float* Ht  = w + OFF_HT;
    float* A   = w + OFF_A;
    float* U   = w + OFF_INV;
    float* Wm  = w + OFF_WM;
    float* G   = w + OFF_G;
    float* J   = w + OFF_J;
    float* X   = w + OFF_X;
    float* Yv  = w + OFF_YV;
    float* Vm  = w + OFF_VM;
    float* AxT = w + OFF_AX;
    float* hid = w + OFF_HID;
    float* cel = w + OFF_CEL;
    float* hv  = w + OFF_HV;
    float* yt  = w + OFF_YT;
    float* uv  = w + OFF_UV;
    float* gb  = w + OFF_GB;
    float* e   = w + OFF_E;
    float* gn2 = w + OFF_GN2;
    float* dgy = w + OFF_DGY;
    float* yn2 = w + OFF_YN2;
    float* sx  = w + OFF_SX;
    float* sx2 = w + OFF_SX2;
    float* ai  = w + OFF_AI;
    float* sdf = w + OFF_SDF;
    float* fm  = w + OFF_FM;
    float* scal = w + OFF_SC;
    _Float16* Jh = (_Float16*)(w + OFF_JH);
    _Float16* xh = (_Float16*)(w + OFF_XH);
    _Float16* sh = (_Float16*)(w + OFF_SH);

    // ---- setup ----
    k_init<<<4096, 256, 0, stream>>>(X, Yv, Vm, hid, cel, h0, c0in, scal);
    k_build_ht<<<256, 256, 0, stream>>>(Hr, Hi, yr, yi, Ht, yt);
    k_ata<<<256, 256, 0, stream>>>(Ht, lam, A);
    k_invert<<<1, 1024, 0, stream>>>(A, U, lam);
    k_gemm256<<<256, 256, 0, stream>>>(U, Ht, Wm, 0);   // W = U@Ht
    k_gemm256<<<256, 256, 0, stream>>>(Ht, Wm, G, 1);   // G = Ht^T@W
    k_uvec<<<1, 256, 0, stream>>>(U, yt, uv);
    k_hvec<<<2, 256, 0, stream>>>(Ht, uv, hv);
    k_buildJ<<<1024, 256, 0, stream>>>(G, J, Jh, &scal[0]);
    k_c0<<<1, 32, 0, stream>>>(scal);

    // ---- initial energy / gbest / feature-normalization stats ----
    k_conv<<<4096, 256, 0, stream>>>(X, xh, sh, e);
    k_gemm_wmma<<<dim3(16, 16), 256, 0, stream>>>(Jh, sh, xh, hv, AxT, e);
    k_argmin<<<1, 1024, 0, stream>>>(e, X, gb, scal);
    k_stats<<<256, 256, 0, stream>>>(X, Yv, gb, gn2, dgy, yn2, sx, sx2);
    k_featstats<<<1, 256, 0, stream>>>(e, gn2, sx, sx2, fm);

    // ---- main loop ----
    for (int i = 0; i < TSTEPS; ++i) {
        float t = (float)i / (float)TSTEPS;
        float p = 0.1f + 0.45f * (1.0f + cosf(3.14159265358979f * t));
        float alpha = 0.9f - 0.4f * t;
        // xh/sh/e already prepared by k_conv (i==0) or previous k_update
        k_gemm_wmma<<<dim3(16, 16), 256, 0, stream>>>(Jh, sh, xh, hv, AxT, e);
        k_argmin<<<1, 1024, 0, stream>>>(e, X, gb, scal);
        k_stats<<<256, 256, 0, stream>>>(X, Yv, gb, gn2, dgy, yn2, sx, sx2);
        k_lstm<<<8, 256, 0, stream>>>(t, e, gn2, dgy, yn2, sx, sx2, fm, hid, cel,
                                      W_ih, W_hh, b_ih, b_hh, W1, b1, W2, b2, ai, sdf);
        k_update<<<4096, 256, 0, stream>>>(i, t, p, alpha, Delta, eta, scal, hv,
                                           gb, AxT, ai, sdf, X, Yv, Vm, xh, sh, e);
    }
    k_out<<<4096, 256, 0, stream>>>(X, (float*)d_out);
}
